// LSTMEncoder_21285857919391
// MI455X (gfx1250) — matrix operand order repaired and checked by reading, not compiled, not run on hardware
//
#include <hip/hip_runtime.h>
#include <hip/hip_bf16.h>

// Shapes
#define B_  16
#define T_  2048
#define D_  512
#define H_  512
#define G4_ 2048   // 4*H
#define O_  256
#define CH_ 256    // recurrent chunk length (T_/CH_ = 8 chunks)

// LDS layout for the recurrent kernel (dynamic shared memory)
#define WHH_ROWS   128                           // gate columns owned by one WG
#define WHH_PITCH  520                           // 512 + 8 bf16 pad
#define WHH_BYTES  (WHH_ROWS * WHH_PITCH * 2)    // 133120
#define GBUF_BYTES (8 * 16 * 16 * 4)             // 8192
#define HPITCH     520                           // h stage pitch (bf16 elems)
#define HST_BYTES  (16 * HPITCH * 2)             // 16640
#define REC_LDS    (WHH_BYTES + GBUF_BYTES + HST_BYTES)  // 157952 (<320KB WGP)

typedef __attribute__((ext_vector_type(16))) __bf16 v16bf;
typedef __attribute__((ext_vector_type(8)))  __bf16 v8bf;
typedef __attribute__((ext_vector_type(8)))  float  v8f;

// ---------------------------------------------------------------------------
// CDNA5 async global->LDS helpers (ASYNCcnt-tracked)
// ---------------------------------------------------------------------------
__device__ __forceinline__ void async_ld16(unsigned ldsoff, const void* g) {
  asm volatile("global_load_async_to_lds_b128 %0, %1, off"
               :: "v"(ldsoff), "v"((unsigned long long)(uintptr_t)g)
               : "memory");
}
template <int N>
__device__ __forceinline__ void wait_async() {
  asm volatile("s_wait_asynccnt %0" :: "i"(N) : "memory");
}

// ---------------------------------------------------------------------------
// WMMA fragment loaders (v_wmma_f32_16x16x32_bf16 layouts, wave32)
// A (16x32 MxK): lane l (m=l&15, hi=l>>4): elems[0..7]=K k0+hi*16..,
//                elems[8..15]=K k0+hi*16+8.. (contiguous, same order as B)
// B (32x16 KxN): col n = n0+(l&15); 16 contiguous K at k0+hi*16 (W row-major
//                [N][K] -> column n of B = row n of W, contiguous loads).
// C/D (16x16 f32): VGPR r, lane l -> m = r + hi*8, n = l&15.
// ---------------------------------------------------------------------------
__device__ __forceinline__ v16bf load_afrag(const __bf16* __restrict__ A,
                                            int lda, int k0, int lane) {
  const int m  = lane & 15;
  const int hi = lane >> 4;
  const __bf16* p = A + m * lda + k0 + hi * 16;
  v8bf c0 = *(const v8bf*)(p);
  v8bf c1 = *(const v8bf*)(p + 8);
  v16bf a;
#pragma unroll
  for (int i = 0; i < 8; ++i) { a[i] = c0[i]; a[8 + i] = c1[i]; }
  return a;
}

// A-fragment from a padded LDS image (ds_load path).
__device__ __forceinline__ v16bf load_afrag_lds(const __bf16* A, int pitch,
                                                int k0, int lane) {
  const int m  = lane & 15;
  const int hi = lane >> 4;
  const __bf16* p = A + m * pitch + k0 + hi * 16;
  v8bf c0 = *(const v8bf*)(p);
  v8bf c1 = *(const v8bf*)(p + 8);
  v16bf a;
#pragma unroll
  for (int i = 0; i < 8; ++i) { a[i] = c0[i]; a[8 + i] = c1[i]; }
  return a;
}

__device__ __forceinline__ v16bf load_bfrag(const __bf16* __restrict__ W,
                                            int ldw, int n0, int k0, int lane) {
  const int n  = n0 + (lane & 15);
  const int hi = lane >> 4;
  return *(const v16bf*)(W + n * ldw + k0 + hi * 16);
}

// B-fragment from a padded LDS tile (rows = gate/output cols, pitch elems).
__device__ __forceinline__ v16bf load_bfrag_lds(const __bf16* Wl, int pitch,
                                                int n0, int k0, int lane) {
  const int n  = n0 + (lane & 15);
  const int hi = lane >> 4;
  const __bf16* p = Wl + n * pitch + k0 + hi * 16;
  v8bf c0 = *(const v8bf*)(p);
  v8bf c1 = *(const v8bf*)(p + 8);
  v16bf b;
#pragma unroll
  for (int i = 0; i < 8; ++i) { b[i] = c0[i]; b[8 + i] = c1[i]; }
  return b;
}

__device__ __forceinline__ float sigmoidf_(float x) {
  return 1.0f / (1.0f + __expf(-x));
}

// ---------------------------------------------------------------------------
__global__ void cvt_bf16_kernel(const float* __restrict__ src,
                                __bf16* __restrict__ dst, int n) {
  int i = blockIdx.x * blockDim.x + threadIdx.x;
  int stride = gridDim.x * blockDim.x;
  for (; i < n; i += stride) dst[i] = (__bf16)src[i];
}

__global__ void bias_add_kernel(const float* __restrict__ a,
                                const float* __restrict__ b,
                                float* __restrict__ o, int n) {
  int i = blockIdx.x * blockDim.x + threadIdx.x;
  if (i < n) o[i] = a[i] + b[i];
}

__global__ void mask_kernel(const float* __restrict__ x,
                            float* __restrict__ mask, int rows) {
  int r = blockIdx.x * blockDim.x + threadIdx.x;
  if (r >= rows) return;
  const float4* p = (const float4*)(x + (size_t)r * D_);
  float s = 0.f;
#pragma unroll 4
  for (int i = 0; i < D_ / 4; ++i) {
    float4 v = p[i];
    s += v.x + v.y + v.z + v.w;
  }
  mask[r] = (s != 0.f) ? 1.f : 0.f;
}

// ---------------------------------------------------------------------------
// Throughput GEMM: GX[b][trel][4H] = X[b, t0+trel, :] @ W.T + bias for one
// chunk of CH_ steps. Grid (32, 16) x 256 threads; block tile 128M x 128N.
// B tile (128 cols x 32 K, shared by all 8 waves) is async-staged into LDS
// with 2-deep double buffering: issue stage k+1, s_wait_asynccnt 2 (tolerates
// the in-flight next stage), barrier, ds-fed WMMAs. A-fragments (per-wave
// private rows, 8x register reuse) stay as direct global clauses.
// ---------------------------------------------------------------------------
#define BST_PITCH 40   // bf16 elems per LDS row (80B -> 4-bank stride)
__global__ __launch_bounds__(256) void gemm_gates_kernel(
    const __bf16* __restrict__ X,    // [B][T][512] bf16
    const __bf16* __restrict__ W,    // [2048][512] bf16
    const float*  __restrict__ bias, // [2048] fused b_ih + b_hh
    float* __restrict__ GX,          // [B][CH_][2048] fp32
    int t0) {
  __shared__ __bf16 Bst[2][128][BST_PITCH];   // 20.5KB double-buffered B tile

  const int tid  = threadIdx.x;
  const int lane = tid & 31;
  const int wave = tid >> 5;
  const int rt    = blockIdx.x * 8 + wave;    // global row-tile 0..255
  const int b     = rt >> 4;                  // 16 row-tiles per batch entry
  const int trel0 = (rt & 15) * 16;
  const __bf16* Arow = X + ((size_t)b * T_ + t0 + trel0) * 512;
  const int n0 = blockIdx.y * 128;

  const unsigned bst_base = (unsigned)(uintptr_t)(void*)&Bst[0][0][0];

  // async-issue one 128x32 B stage (2 instructions per thread -> asynccnt 2)
  auto issue_stage = [&](int kt) {
    const int buf = kt & 1;
    const int k0  = kt * 32;
#pragma unroll
    for (int i = 0; i < 2; ++i) {
      const int ci   = tid * 2 + i;        // 0..511 16B chunks
      const int row  = ci >> 2;            // B-tile row (output col) 0..127
      const int part = ci & 3;             // 16B piece within 64B row
      const __bf16* src = W + (size_t)(n0 + row) * 512 + k0 + part * 8;
      const unsigned ldsoff =
          bst_base + (unsigned)(((buf * 128 + row) * BST_PITCH + part * 8) * 2);
      async_ld16(ldsoff, src);
    }
  };

  v8f acc[8];
#pragma unroll
  for (int j = 0; j < 8; ++j) acc[j] = (v8f){};

  issue_stage(0);
  for (int kt = 0; kt < 16; ++kt) {
    if (kt < 15) { issue_stage(kt + 1); wait_async<2>(); }
    else         { wait_async<0>(); }
    __syncthreads();                       // stage kt visible to all waves
    const __bf16* Bl = &Bst[kt & 1][0][0];
    v16bf a = load_afrag(Arow, 512, kt * 32, lane);
#pragma unroll
    for (int j = 0; j < 8; ++j) {
      v16bf bb = load_bfrag_lds(Bl, BST_PITCH, j * 16, 0, lane);
      acc[j] = __builtin_amdgcn_wmma_f32_16x16x32_bf16(false, a, false, bb,
                                                       (short)0, acc[j],
                                                       false, false);
    }
    __syncthreads();                       // reads done before overwrite
  }

  const int nn = lane & 15, hi = lane >> 4;
#pragma unroll
  for (int j = 0; j < 8; ++j) {
    const int n = n0 + j * 16 + nn;
    const float bv = bias[n];
#pragma unroll
    for (int r = 0; r < 8; ++r) {
      const int trel = trel0 + r + hi * 8;
      GX[((size_t)b * CH_ + trel) * G4_ + n] = acc[j][r] + bv;
    }
  }
}

// ---------------------------------------------------------------------------
// Persistent recurrent kernel for one chunk [t0, t0+CH_).
// 16 WGs x 256 threads; WG wg owns h columns [wg*32, wg*32+32).
// Whh slice async-DMA'd to LDS once; per step: GX accumulator loads overlap
// the barrier spin, h image async-staged to LDS (one coalesced 16KB transfer
// instead of 8 redundant per-wave L2 pulls on the serial path), 16 LDS-fed
// WMMAs, LDS gate exchange, fused masked update, arrive-only barrier.
// ---------------------------------------------------------------------------
__global__ __launch_bounds__(256) void lstm_rec_kernel(
    const float*  __restrict__ GX,      // [B][CH_][2048] x-gates + bias
    const __bf16* __restrict__ Whh,     // [2048][512] bf16
    const float*  __restrict__ mask,    // [B*T]
    __bf16* __restrict__ hbuf,          // [2][16][512] bf16
    __bf16* __restrict__ seq_out,       // [B][T][512] bf16 or nullptr
    unsigned* __restrict__ barrier_cnt, // monotonic across chunks
    float* __restrict__ cstate,         // [16][512] fp32 persistent cell
    int t0) {
  extern __shared__ char smem[];
  __bf16* whhl = (__bf16*)smem;
  float (*gtile)[16][16] = (float (*)[16][16])(smem + WHH_BYTES);
  __bf16* hl = (__bf16*)(smem + WHH_BYTES + GBUF_BYTES);

  const int wg   = blockIdx.x;
  const int tid  = threadIdx.x;
  const int lane = tid & 31;
  const int wave = tid >> 5;
  const int gsel = wave >> 1;                    // gate 0..3 (i,f,g,o)
  const int nt   = wave & 1;
  const int col0 = wg * 32;
  const int ncol = gsel * 512 + col0 + nt * 16;  // gate column base (global)
  const int lrow = gsel * 32 + nt * 16;          // LDS Whh local row base

  const unsigned whh_base = (unsigned)(uintptr_t)(void*)whhl;
  const unsigned hl_base  = (unsigned)(uintptr_t)(void*)hl;

  // ---- one-time: async DMA the WG's 128KB Whh slice into LDS ----
  for (int ci = tid; ci < (WHH_ROWS * 512 * 2) / 16; ci += 256) {
    const int byteoff = ci * 16;            // within unpadded 1KB rows
    const int row     = byteoff >> 10;      // local gate col 0..127
    const int inrow   = byteoff & 1023;
    const int g = row >> 5, c = row & 31;
    const __bf16* gsrc =
        Whh + ((size_t)(g * 512 + col0 + c) * 512) + (inrow >> 1);
    async_ld16(whh_base + (unsigned)(row * WHH_PITCH * 2 + inrow), gsrc);
  }
  wait_async<0>();
  __syncthreads();

  // ---- cell state slice -> registers (2 elements per thread) ----
  float creg[2];
#pragma unroll
  for (int i = 0; i < 2; ++i) {
    const int e = tid + i * 256;
    creg[i] = cstate[(e >> 5) * 512 + col0 + (e & 31)];
  }

  for (int t = t0; t < t0 + CH_; ++t) {
    // acc <- precomputed x-gates (independent of h; overlaps the spin below)
    v8f acc;
    {
      const int nn = lane & 15, hi = lane >> 4;
#pragma unroll
      for (int r = 0; r < 8; ++r) {
        const int b = r + hi * 8;
        acc[r] = GX[((size_t)b * CH_ + (t - t0)) * G4_ + ncol + nn];
      }
    }
    // wait until every WG finished step t-1 (monotonic arrival counter)
    if (tid == 0) {
      const unsigned target = (unsigned)(16 * t);
      volatile unsigned* vb = (volatile unsigned*)barrier_cnt;
      while (*vb < target) __builtin_amdgcn_s_sleep(1);
    }
    __syncthreads();
    __threadfence();   // invalidate so we observe other WGs' h writes

    const __bf16* __restrict__ hread  = hbuf + (size_t)(t & 1) * (16 * 512);
    __bf16* __restrict__       hwrite = hbuf + (size_t)((t + 1) & 1) * (16 * 512);

    // stage full h image (16x512 bf16 = 16KB) into LDS, one coalesced pass
#pragma unroll
    for (int i = 0; i < 4; ++i) {
      const int ci   = tid * 4 + i;     // 0..1023 16B chunks
      const int row  = ci >> 6;         // batch row 0..15
      const int part = ci & 63;         // 16B piece within 1KB row
      const __bf16* src = hread + row * 512 + part * 8;
      async_ld16(hl_base + (unsigned)((row * HPITCH + part * 8) * 2), src);
    }
    wait_async<0>();
    __syncthreads();

    // h_{t-1} @ Whh.T  (A and B fragments both from LDS)
#pragma unroll 4
    for (int k0 = 0; k0 < 512; k0 += 32) {
      v16bf a = load_afrag_lds(hl, HPITCH, k0, lane);
      v16bf b = load_bfrag_lds(whhl, WHH_PITCH, lrow, k0, lane);
      acc = __builtin_amdgcn_wmma_f32_16x16x32_bf16(false, a, false, b,
                                                    (short)0, acc, false, false);
    }

    // exchange gate tiles through LDS
    {
      const int n16 = lane & 15, hi = lane >> 4;
#pragma unroll
      for (int r = 0; r < 8; ++r) gtile[wave][r + hi * 8][n16] = acc[r];
    }
    __syncthreads();

    // fused activations + masked state update (512 elems / 256 threads)
#pragma unroll
    for (int i = 0; i < 2; ++i) {
      const int e   = tid + i * 256;
      const int b   = e >> 5;
      const int col = e & 31;
      const int tl  = col >> 4;
      const int c16 = col & 15;
      float iv = sigmoidf_(gtile[0 * 2 + tl][b][c16]);
      float fv = sigmoidf_(gtile[1 * 2 + tl][b][c16]);
      float gv = tanhf(gtile[2 * 2 + tl][b][c16]);
      float ov = sigmoidf_(gtile[3 * 2 + tl][b][c16]);
      const float cold = creg[i];
      const float cnew = fv * cold + iv * gv;
      const float hnew = ov * tanhf(cnew);
      const bool  m    = mask[b * T_ + t] != 0.f;
      const float hold = (float)hl[b * HPITCH + col0 + col];
      const float hsel = m ? hnew : hold;
      creg[i] = m ? cnew : cold;
      hwrite[b * 512 + col0 + col] = (__bf16)hsel;
      if (seq_out) seq_out[((size_t)b * T_ + t) * 512 + col0 + col] = (__bf16)hsel;
    }

    // publish + arrive (no wait here; the wait is at the top of next step)
    __threadfence();
    __syncthreads();
    if (tid == 0) atomicAdd(barrier_cnt, 1u);
  }

  // persist cell state for the next chunk
#pragma unroll
  for (int i = 0; i < 2; ++i) {
    const int e = tid + i * 256;
    cstate[(e >> 5) * 512 + col0 + (e & 31)] = creg[i];
  }
}

// ---------------------------------------------------------------------------
// Final FC: out[16,256] = hT[16,512] @ Wfc[256,512].T + b_fc
// ---------------------------------------------------------------------------
__global__ __launch_bounds__(256) void fc_kernel(
    const __bf16* __restrict__ h,    // [16][512] bf16 (final hidden)
    const __bf16* __restrict__ Wfc,  // [256][512] bf16
    const float*  __restrict__ bfc,  // [256]
    float* __restrict__ out) {       // [16][256]
  const int lane = threadIdx.x & 31;
  const int wave = threadIdx.x >> 5;
#pragma unroll
  for (int tnum = 0; tnum < 2; ++tnum) {
    const int n0 = (wave * 2 + tnum) * 16;
    v8f acc = {};
#pragma unroll 4
    for (int k0 = 0; k0 < 512; k0 += 32) {
      v16bf a = load_afrag(h, 512, k0, lane);
      v16bf b = load_bfrag(Wfc, 512, n0, k0, lane);
      acc = __builtin_amdgcn_wmma_f32_16x16x32_bf16(false, a, false, b,
                                                    (short)0, acc, false, false);
    }
    const int n = n0 + (lane & 15), hi = lane >> 4;
#pragma unroll
    for (int r = 0; r < 8; ++r) out[(r + hi * 8) * O_ + n] = acc[r] + bfc[n];
  }
}

// ---------------------------------------------------------------------------
extern "C" void kernel_launch(void* const* d_in, const int* in_sizes, int n_in,
                              void* d_out, int out_size, void* d_ws, size_t ws_size,
                              hipStream_t stream) {
  const float* x    = (const float*)d_in[0];
  const float* Wih0 = (const float*)d_in[1];
  const float* Whh0 = (const float*)d_in[2];
  const float* bih0 = (const float*)d_in[3];
  const float* bhh0 = (const float*)d_in[4];
  const float* Wih1 = (const float*)d_in[5];
  const float* Whh1 = (const float*)d_in[6];
  const float* bih1 = (const float*)d_in[7];
  const float* bhh1 = (const float*)d_in[8];
  const float* Wfc  = (const float*)d_in[9];
  const float* bfc  = (const float*)d_in[10];
  float* out = (float*)d_out;

  // ---- workspace carve (256B aligned) ----
  char* ws = (char*)d_ws;
  size_t off = 0;
  auto alloc = [&](size_t bytes) -> void* {
    void* p = ws + off;
    off = (off + bytes + 255) & ~(size_t)255;
    return p;
  };
  const size_t nX = (size_t)B_ * T_ * D_;
  const size_t nW = (size_t)G4_ * D_;
  __bf16* X16    = (__bf16*)alloc(nX * 2);
  __bf16* SEQ1   = (__bf16*)alloc(nX * 2);
  __bf16* Wih0b  = (__bf16*)alloc(nW * 2);
  __bf16* Whh0b  = (__bf16*)alloc(nW * 2);
  __bf16* Wih1b  = (__bf16*)alloc(nW * 2);
  __bf16* Whh1b  = (__bf16*)alloc(nW * 2);
  __bf16* Wfcb   = (__bf16*)alloc((size_t)O_ * H_ * 2);
  float*  bias0  = (float*)alloc(G4_ * 4);
  float*  bias1  = (float*)alloc(G4_ * 4);
  float*  maskb  = (float*)alloc((size_t)B_ * T_ * 4);
  float*  GX     = (float*)alloc((size_t)B_ * CH_ * G4_ * 4);  // 33.5MB chunk
  __bf16* hbufA  = (__bf16*)alloc(2 * 16 * 512 * 2);
  __bf16* hbufB  = (__bf16*)alloc(2 * 16 * 512 * 2);
  float*  cst0   = (float*)alloc(16 * 512 * 4);
  float*  cst1   = (float*)alloc(16 * 512 * 4);
  unsigned* bar0 = (unsigned*)alloc(256);
  unsigned* bar1 = (unsigned*)alloc(256);
  (void)ws_size; (void)n_in; (void)in_sizes; (void)out_size;

  // ---- phase 0: conversions, fused biases, mask, state init ----
  cvt_bf16_kernel<<<2048, 256, 0, stream>>>(x, X16, (int)nX);
  cvt_bf16_kernel<<<1024, 256, 0, stream>>>(Wih0, Wih0b, (int)nW);
  cvt_bf16_kernel<<<1024, 256, 0, stream>>>(Whh0, Whh0b, (int)nW);
  cvt_bf16_kernel<<<1024, 256, 0, stream>>>(Wih1, Wih1b, (int)nW);
  cvt_bf16_kernel<<<1024, 256, 0, stream>>>(Whh1, Whh1b, (int)nW);
  cvt_bf16_kernel<<<256, 256, 0, stream>>>(Wfc, Wfcb, O_ * H_);
  bias_add_kernel<<<8, 256, 0, stream>>>(bih0, bhh0, bias0, G4_);
  bias_add_kernel<<<8, 256, 0, stream>>>(bih1, bhh1, bias1, G4_);
  mask_kernel<<<(B_ * T_ + 255) / 256, 256, 0, stream>>>(x, maskb, B_ * T_);

  hipMemsetAsync(hbufA, 0, 2 * 16 * 512 * 2, stream);
  hipMemsetAsync(hbufB, 0, 2 * 16 * 512 * 2, stream);
  hipMemsetAsync(cst0, 0, 16 * 512 * 4, stream);
  hipMemsetAsync(cst1, 0, 16 * 512 * 4, stream);
  hipMemsetAsync(bar0, 0, 256, stream);
  hipMemsetAsync(bar1, 0, 256, stream);

  // ---- layer 0: alternate x-gates GEMM and recurrent chunk ----
  for (int c = 0; c < T_ / CH_; ++c) {
    gemm_gates_kernel<<<dim3(32, 16), 256, 0, stream>>>(X16, Wih0b, bias0, GX,
                                                        c * CH_);
    lstm_rec_kernel<<<16, 256, REC_LDS, stream>>>(GX, Whh0b, maskb, hbufA,
                                                  SEQ1, bar0, cst0, c * CH_);
  }
  // ---- layer 1 ----
  for (int c = 0; c < T_ / CH_; ++c) {
    gemm_gates_kernel<<<dim3(32, 16), 256, 0, stream>>>(SEQ1, Wih1b, bias1, GX,
                                                        c * CH_);
    lstm_rec_kernel<<<16, 256, REC_LDS, stream>>>(GX, Whh1b, maskb, hbufB,
                                                  nullptr, bar1, cst1, c * CH_);
  }

  // T even -> final h of layer 1 sits in buffer 0 of hbufB
  fc_kernel<<<1, 256, 0, stream>>>(hbufB, Wfcb, bfc, out);
}
